// InverseRadonTransform_1108101562469
// MI455X (gfx1250) — matrix operand order repaired and checked by reading, not compiled, not run on hardware
//
#include <hip/hip_runtime.h>
#include <hip/hip_bf16.h>

// ---------------------------------------------------------------------------
// Inverse Radon (filtered backprojection), MI455X / gfx1250.
//  K1: build spatial-domain Shepp-Logan filter hr[512] via direct f64 DFTs.
//  K2: filtered = S(720x512) @ circulant(hr)  as f16 WMMA GEMM, f32 accum.
//  K3: backprojection, 30-angle LDS-staged chunks, bilinear taps from LDS.
// ---------------------------------------------------------------------------

typedef __attribute__((ext_vector_type(16))) _Float16 v16h;
typedef __attribute__((ext_vector_type(8)))  _Float16 v8h;
typedef __attribute__((ext_vector_type(8)))  float    v8f;

#define FBP_P      512
#define FBP_A      180
#define FBP_B      4
#define FBP_ROWS   (FBP_B * FBP_A)   // 720
#define SROW       520               // padded f16 row stride (1040B, 16B-mult)
#define ACH        30                // angles staged per LDS chunk

static constexpr double PI_D = 3.14159265358979323846;

// ---------------------------------------------------------------------------
// K1: hr[n] = real(ifft(ff))[n], ff = shepp-logan frequency response.
// One block of 512 threads; f64 math matches the numpy float64 reference.
// ---------------------------------------------------------------------------
__global__ __launch_bounds__(512) void fbp_make_filter(float* __restrict__ hr) {
    __shared__ double ffs[FBP_P];
    const int m = threadIdx.x;

    // ff[m] = 2 * sum_k f[k] * cos(2*pi*m*k/512)
    double acc = 2.0 * 0.25;                       // k = 0 term
    const double w = 2.0 * PI_D * (double)m / (double)FBP_P;
    for (int j = 0; j < 256; ++j) {
        const int k  = 2 * j + 1;
        const int nk = (j < 128) ? (2 * j + 1) : (511 - 2 * j);
        const double fk = -1.0 / ((PI_D * nk) * (PI_D * nk));
        acc += 2.0 * fk * cos(w * (double)k);
    }
    if (m > 0) {                                   // ff[1:] *= sin(om)/om
        const double freq = ((m < 256) ? (double)m : (double)(m - 512)) / 512.0;
        const double om = PI_D * freq;
        acc *= sin(om) / om;
    }
    ffs[m] = acc;
    __syncthreads();

    // hr[n] = (1/512) * sum_m ff[m] * cos(2*pi*m*n/512)
    const int n = threadIdx.x;
    const double wn = 2.0 * PI_D * (double)n / (double)FBP_P;
    double h = 0.0;
    for (int mm = 0; mm < FBP_P; ++mm) h += ffs[mm] * cos(wn * (double)mm);
    hr[n] = (float)(h / (double)FBP_P);
}

// ---------------------------------------------------------------------------
// K2: filtered[r][n] = sum_k sino[r][k] * hr[(n-k) & 511]
// 256 threads = 8 waves; each wave owns one 16x16 output tile,
// block covers 16 rows x 128 cols.  v_wmma_f32_16x16x32_f16, K=512 in 16 steps.
// ---------------------------------------------------------------------------
__global__ __launch_bounds__(256) void fbp_filter_gemm(
    const float* __restrict__ sino,   // [720][512]
    const float* __restrict__ hr,     // [512]
    float* __restrict__ filt)         // [720][512]
{
    __shared__ _Float16 sA[16 * SROW];   // 16 f16 sinogram rows, padded
    __shared__ _Float16 rev[1024];       // rev[i] = hr[(1023 - i) & 511]

    const int tid = threadIdx.x;
    const int r0  = blockIdx.x * 16;

    // Stage 16 rows, f32 -> f16, vectorized 16B global loads.
    const float4* src = (const float4*)(sino + (size_t)r0 * FBP_P);
    #pragma unroll
    for (int t = 0; t < 8; ++t) {
        const int i4  = tid + t * 256;     // 0..2047 float4s
        const int row = i4 >> 7;
        const int c4  = i4 & 127;
        const float4 v = src[row * 128 + c4];
        _Float16* dst = &sA[row * SROW + c4 * 4];
        dst[0] = (_Float16)v.x; dst[1] = (_Float16)v.y;
        dst[2] = (_Float16)v.z; dst[3] = (_Float16)v.w;
    }
    // Reversed-duplicated circulant table: Hc[k][n] == rev[511 - n + k].
    #pragma unroll
    for (int t = 0; t < 4; ++t) {
        const int i = tid + t * 256;
        rev[i] = (_Float16)hr[(1023 - i) & 511];
    }
    __syncthreads();

    const int lane  = tid & 31;
    const int wave  = tid >> 5;
    const int n0    = (blockIdx.y * 8 + wave) * 16;  // output col tile base
    const int m     = lane & 15;                     // A row (both half-waves)
    const int khalf = lane >> 4;                     // K half select
    const int nB    = lane & 15;                     // B col

    v8f c = {};
    for (int k0 = 0; k0 < FBP_P; k0 += 32) {
        // A fragment: lane holds M=m, K = {khalf*8 + 0..7, 16+khalf*8 + 0..7}.
        // Both runs are contiguous, 16B-aligned in LDS -> two b128 ds loads.
        const v8h* pa = (const v8h*)&sA[m * SROW + k0 + khalf * 8];
        const v8h lo = pa[0];
        const v8h hi = pa[2];                        // +16 halves
        const v16h a = __builtin_shufflevector(lo, hi,
            0,1,2,3,4,5,6,7, 8,9,10,11,12,13,14,15);

        // B fragment: lane holds N=nB, K = {khalf*8 + 0..7, 16+khalf*8 + 0..7}.
        // b[j] = Hc[k0+khalf*8+j+(j>=8?8:0)][n0+nB] = rev[base + j + (j>=8?8:0)].
        const int base = 511 - (n0 + nB) + k0 + khalf * 8;
        v16h b;
        #pragma unroll
        for (int j = 0; j < 16; ++j) b[j] = rev[base + j + (j >= 8 ? 8 : 0)];

        c = __builtin_amdgcn_wmma_f32_16x16x32_f16(
                /*neg_a=*/false, a, /*neg_b=*/false, b,
                /*c_mod=*/(short)0, c, /*reuse_a=*/false, /*reuse_b=*/false);
    }

    // C/D layout: lanes 0-15 -> M = v, lanes 16-31 -> M = 8 + v; N = lane&15.
    const int col  = n0 + (lane & 15);
    const int mrow = (lane >> 4) * 8;
    #pragma unroll
    for (int v = 0; v < 8; ++v)
        filt[(size_t)(r0 + mrow + v) * FBP_P + col] = c[v];
}

// ---------------------------------------------------------------------------
// K3: backprojection. One pixel per thread, 16x16 tile per block, one batch
// per blockIdx.z. Projection rows staged in LDS, 30 angles per chunk (60KB).
// ---------------------------------------------------------------------------
__global__ __launch_bounds__(256) void fbp_backproject(
    const float* __restrict__ filt,   // [B][A][P]
    float* __restrict__ out)          // [B][512][512]
{
    __shared__ float sProj[ACH * FBP_P];
    __shared__ float sSin[ACH];
    __shared__ float sCos[ACH];

    const int tid = threadIdx.x;
    const int x = blockIdx.x * 16 + (tid & 15);
    const int y = blockIdx.y * 16 + (tid >> 4);
    const int b = blockIdx.z;

    const float xv = 255.0f - (float)x;   // reversed x grid, minus S/2
    const float yv = (float)y - 256.0f;

    float acc = 0.0f;
    for (int a0 = 0; a0 < FBP_A; a0 += ACH) {
        __syncthreads();
        // Stage ACH contiguous projection rows (30*512 f32 = 3840 float4).
        const float4* src = (const float4*)(filt + ((size_t)b * FBP_A + a0) * FBP_P);
        float4* dst = (float4*)sProj;
        #pragma unroll
        for (int t = 0; t < (ACH * 128) / 256; ++t)
            dst[tid + t * 256] = src[tid + t * 256];
        if (tid < ACH) {
            // angles_rad[a] = a * (180/179 deg) = a * pi/179
            const float ang = (float)(a0 + tid) * (3.14159265358979f / 179.0f);
            sSin[tid] = sinf(ang);
            sCos[tid] = cosf(ang);
        }
        __syncthreads();

        #pragma unroll 5
        for (int aa = 0; aa < ACH; ++aa) {
            const float p  = fmaf(xv, sSin[aa], fmaf(yv, sCos[aa], 256.0f));
            const float pf = floorf(p);
            const int   i0 = (int)pf;
            const float fr = p - pf;
            const int   i1 = i0 + 1;
            const float v0 = ((unsigned)i0 < (unsigned)FBP_P) ? sProj[aa * FBP_P + i0] : 0.0f;
            const float v1 = ((unsigned)i1 < (unsigned)FBP_P) ? sProj[aa * FBP_P + i1] : 0.0f;
            acc = fmaf(1.0f - fr, v0, fmaf(fr, v1, acc));
        }
    }

    const float rmask = (fmaf(xv, xv, yv * yv) <= 65536.0f) ? 1.0f : 0.0f;
    out[((size_t)b * 512 + y) * 512 + x] = acc * (3.14159265358979f / 360.0f) * rmask;
}

// ---------------------------------------------------------------------------
extern "C" void kernel_launch(void* const* d_in, const int* in_sizes, int n_in,
                              void* d_out, int out_size, void* d_ws, size_t ws_size,
                              hipStream_t stream) {
    (void)in_sizes; (void)n_in; (void)out_size; (void)ws_size;

    const float* sino = (const float*)d_in[0];     // [4][180][512] f32
    float* out  = (float*)d_out;                   // [4][1][512][512] f32

    float* hr   = (float*)d_ws;                    // 512 f32
    float* filt = hr + FBP_P;                      // 720*512 f32 (16B aligned)

    fbp_make_filter<<<1, 512, 0, stream>>>(hr);

    fbp_filter_gemm<<<dim3(FBP_ROWS / 16, FBP_P / 128), 256, 0, stream>>>(
        sino, hr, filt);

    fbp_backproject<<<dim3(32, 32, FBP_B), 256, 0, stream>>>(filt, out);
}
